// ConfidenceGatedModel_35270271435249
// MI455X (gfx1250) — matrix operand order repaired and checked by reading, not compiled, not run on hardware
//
#include <hip/hip_runtime.h>
#include <hip/hip_bf16.h>

// ---------------- types ----------------
typedef __attribute__((ext_vector_type(16))) __bf16       v16bf;
typedef __attribute__((ext_vector_type(8)))  float        v8f;
typedef __attribute__((ext_vector_type(8)))  unsigned int v8u;
typedef __attribute__((ext_vector_type(4)))  unsigned int v4u;

// ---------------- problem constants (from reference) ----------------
#define B_    32
#define T_    2048
#define H_    1024
#define H2_   2048
#define V_    32000
#define NC_   2045      // n_cand = T - 3
#define KSEL  128       // MEMORY_SLOTS
#define QTOK  2046      // query position T - 2
#define LNEPS 1e-5f

// padded LDS strides (bank-conflict-free for 16 consecutive rows)
#define SH_STRIDE 1028  // floats per h/hidden row
#define SF_STRIDE 2056  // bf16 per ff1 row (4112B)

// ---------------- wave helpers (wave32) ----------------
__device__ __forceinline__ float wave_sum(float v) {
#pragma unroll
  for (int o = 16; o > 0; o >>= 1) v += __shfl_xor(v, o, 32);
  return v;
}
__device__ __forceinline__ float wave_max(float v) {
#pragma unroll
  for (int o = 16; o > 0; o >>= 1) v = fmaxf(v, __shfl_xor(v, o, 32));
  return v;
}

// ---------------- WMMA fragment loaders ----------------
// B (32x16 bf16) fragment: lane n (<16) holds K=0..7 and 16..23 of col n, lane n+16 holds K=8..15 and 24..31.
// With weights stored column-contiguous (K-major), this is two 16B loads.
__device__ __forceinline__ v16bf load_b_frag(const __bf16* p) {
  union { v16bf h; v4u u[2]; } f;
  f.u[0] = *(const v4u*)p;
  f.u[1] = *(const v4u*)(p + 16);
  return f.h;
}
// A (16x32 bf16) fragment from a contiguous bf16 LDS row.
// lane<16: K = 0..7 (vgpr0-3) and 16..23 (vgpr4-7); lane>=16: +8.
__device__ __forceinline__ v16bf load_a_frag_bf16(const __bf16* row, int k, int lane) {
  const int kb = (lane < 16) ? 0 : 8;
  union { v16bf h; v4u u[2]; } f;
  f.u[0] = *(const v4u*)(row + k + kb);        // K = k+kb .. +7
  f.u[1] = *(const v4u*)(row + k + 16 + kb);   // K = k+16+kb .. +7
  return f.h;
}
// A fragment from an f32 LDS row (convert to bf16 on the fly).
__device__ __forceinline__ v16bf load_a_frag_f32(const float* row, int k, int lane) {
  const int kb = (lane < 16) ? 0 : 8;
  v16bf a;
#pragma unroll
  for (int i = 0; i < 8; ++i) {
    a[i]     = (__bf16)row[k + kb + i];
    a[i + 8] = (__bf16)row[k + 16 + kb + i];
  }
  return a;
}

// ---------------- weight convert + transpose (f32 row-major -> bf16 col-major) ----------------
__global__ __launch_bounds__(256)
void k_cvt_transpose(const float* __restrict__ src, __bf16* __restrict__ dst,
                     int rows, int cols) {
  size_t n = (size_t)rows * cols;
  for (size_t i = (size_t)blockIdx.x * 256 + threadIdx.x; i < n;
       i += (size_t)gridDim.x * 256) {
    int r = (int)(i / cols);
    int c = (int)(i % cols);
    dst[(size_t)c * rows + r] = (__bf16)src[i];
  }
}

// ---------------- fused FFN + LayerNorm kernel ----------------
// mode 0: per-token gating scores for tokens [0, NC_)
// mode 1: hidden rows for the 128 selected tokens + the query token
//
// Schedule: k-outer / tile-inner. The A fragment is built once per k-step and
// reused by all live accumulators, so WMMA issue dominates (A-frag cost /16).
__global__ __launch_bounds__(256)
void k_ffn(int mode,
           const int*   __restrict__ seq,  const float* __restrict__ embed,
           const __bf16* __restrict__ w1t, const float* __restrict__ b1,
           const __bf16* __restrict__ w2t, const float* __restrict__ b2,
           const float* __restrict__ ln_g, const float* __restrict__ ln_b,
           const float* __restrict__ wg_w, const float* __restrict__ wg_b,
           const int*   __restrict__ top_idx,
           float* __restrict__ scores, float* __restrict__ memv,
           float* __restrict__ qh) {
  extern __shared__ char smraw[];
  float*  sH   = (float*)smraw;                                   // 16 x SH_STRIDE f32
  __bf16* sF   = (__bf16*)(smraw + 16 * SH_STRIDE * 4);           // 16 x SF_STRIDE bf16
  int*    sTok = (int*)(smraw + 16 * SH_STRIDE * 4 + 16 * SF_STRIDE * 2);

  const int tid  = threadIdx.x;
  const int lane = tid & 31;
  const int wv   = tid >> 5;          // 8 waves
  const int tile = blockIdx.x;
  const int b    = blockIdx.y;

  if (tid < 16) {
    int tok;
    if (mode == 0) {
      tok = tile * 16 + tid;
      if (tok >= NC_) tok = NC_ - 1;  // padded rows: computed, never written
    } else {
      int g = tile * 16 + tid;
      tok = (g < KSEL) ? top_idx[b * KSEL + g] : QTOK;  // g==128 -> query; g>128 pad
    }
    sTok[tid] = tok;
  }
  __syncthreads();

  // gather embedding rows into f32 LDS (coalesced)
  for (int i = tid; i < 16 * H_; i += 256) {
    int r = i >> 10, c = i & (H_ - 1);
    sH[r * SH_STRIDE + c] =
        embed[(size_t)seq[(size_t)b * T_ + sTok[r]] * H_ + c];
  }
  __syncthreads();

  const int n16 = lane & 15;
  const int m0  = (lane < 16) ? 0 : 8;  // C/D: vgpr j holds row m0+j, col n16
  const int ks  = (lane < 16) ? 0 : 8;  // B-frag K offset

  // ---- Phase A: ff1 = relu(h @ w1 + b1)  -> sF (bf16) ----
  // wave wv owns 16 contiguous N-tiles: n0 = wv*256 + i*16, i = 0..15
  {
    v8f acc[16];
#pragma unroll
    for (int i = 0; i < 16; ++i) {
      const float bias = b1[wv * 256 + i * 16 + n16];
      acc[i] = (v8f){bias, bias, bias, bias, bias, bias, bias, bias};
    }
    const __bf16* colp = w1t + (size_t)(wv * 256 + n16) * H_ + ks;
    const float*  arow = sH + (size_t)n16 * SH_STRIDE;
    for (int k = 0; k < H_; k += 32) {
      v16bf a = load_a_frag_f32(arow, k, lane);          // built once per k-step
      if (k + 32 < H_) __builtin_prefetch(colp + k + 32, 0, 1);
#pragma unroll
      for (int i = 0; i < 16; ++i) {
        v16bf bb = load_b_frag(colp + (size_t)i * 16 * H_ + k);
        acc[i] = __builtin_amdgcn_wmma_f32_16x16x32_bf16(false, a, false, bb,
                                                         (short)0, acc[i], false, false);
      }
    }
#pragma unroll
    for (int i = 0; i < 16; ++i) {
      const int n0 = wv * 256 + i * 16;
#pragma unroll
      for (int j = 0; j < 8; ++j)
        sF[(m0 + j) * SF_STRIDE + n0 + n16] = (__bf16)fmaxf(acc[i][j], 0.0f);
    }
  }
  __syncthreads();

  // ---- Phase B: hidden_pre = h + ff1 @ w2 + b2 (accumulate into sH) ----
  // wave wv owns 8 contiguous N-tiles: n0 = wv*128 + i*16, i = 0..7
  {
    v8f acc[8];
#pragma unroll
    for (int i = 0; i < 8; ++i) {
      const float bias = b2[wv * 128 + i * 16 + n16];
      acc[i] = (v8f){bias, bias, bias, bias, bias, bias, bias, bias};
    }
    const __bf16* colp = w2t + (size_t)(wv * 128 + n16) * H2_ + ks;
    const __bf16* arow = sF + (size_t)n16 * SF_STRIDE;
    for (int k = 0; k < H2_; k += 32) {
      v16bf a = load_a_frag_bf16(arow, k, lane);         // built once per k-step
      if (k + 32 < H2_) __builtin_prefetch(colp + k + 32, 0, 1);
#pragma unroll
      for (int i = 0; i < 8; ++i) {
        v16bf bb = load_b_frag(colp + (size_t)i * 16 * H2_ + k);
        acc[i] = __builtin_amdgcn_wmma_f32_16x16x32_bf16(false, a, false, bb,
                                                         (short)0, acc[i], false, false);
      }
    }
#pragma unroll
    for (int i = 0; i < 8; ++i) {
      const int n0 = wv * 128 + i * 16;
#pragma unroll
      for (int j = 0; j < 8; ++j)
        sH[(m0 + j) * SH_STRIDE + n0 + n16] += acc[i][j];  // unique owner per element
    }
  }
  __syncthreads();

  // ---- LayerNorm + output (2 rows per wave) ----
  for (int rr = 0; rr < 2; ++rr) {
    const int r = wv * 2 + rr;
    float* row = sH + (size_t)r * SH_STRIDE;
    float s = 0.f;
    for (int c = lane; c < H_; c += 32) s += row[c];
    const float mean = wave_sum(s) * (1.0f / H_);
    float v2 = 0.f;
    for (int c = lane; c < H_; c += 32) { float d = row[c] - mean; v2 += d * d; }
    const float rstd = rsqrtf(wave_sum(v2) * (1.0f / H_) + LNEPS);

    if (mode == 0) {
      float sc = 0.f;
      for (int c = lane; c < H_; c += 32) {
        float nv = (row[c] - mean) * rstd * ln_g[c] + ln_b[c];
        sc += nv * wg_w[c];
      }
      sc = wave_sum(sc);
      const int tokG = tile * 16 + r;
      if (lane == 0 && tokG < NC_)
        scores[(size_t)b * 2048 + tokG] = sc + wg_b[0];
    } else {
      const int g = tile * 16 + r;
      float* dst = nullptr;
      if (g < KSEL)       dst = memv + ((size_t)b * KSEL + g) * H_;
      else if (g == KSEL) dst = qh + (size_t)b * H_;
      if (dst)
        for (int c = lane; c < H_; c += 32)
          dst[c] = (row[c] - mean) * rstd * ln_g[c] + ln_b[c];
    }
  }
}

// ---------------- top-128 per batch (iterative argmax, then index sort) ----------------
__global__ __launch_bounds__(256)
void k_topk(const float* __restrict__ scores, int* __restrict__ top_idx) {
  const int b = blockIdx.x, tid = threadIdx.x;
  extern __shared__ char smraw[];
  float* sv  = (float*)smraw;          // 2048
  int*   sel = (int*)(sv + 2048);      // 128
  float* rv  = (float*)(sel + 128);    // 256
  int*   ri  = (int*)(rv + 256);       // 256

  for (int i = tid; i < 2048; i += 256)
    sv[i] = (i < NC_) ? scores[(size_t)b * 2048 + i] : -3.0e38f;
  __syncthreads();

  for (int s = 0; s < KSEL; ++s) {
    float best = -3.0e38f; int bi = 0x7fffffff;
    for (int i = tid; i < NC_; i += 256) {
      float v = sv[i];
      if (v > best) { best = v; bi = i; }   // first hit => smallest index on ties
    }
    rv[tid] = best; ri[tid] = bi;
    __syncthreads();
    for (int off = 128; off > 0; off >>= 1) {
      if (tid < off) {
        float vo = rv[tid + off]; int io = ri[tid + off];
        if (vo > rv[tid] || (vo == rv[tid] && io < ri[tid])) { rv[tid] = vo; ri[tid] = io; }
      }
      __syncthreads();
    }
    if (tid == 0) { sel[s] = ri[0]; sv[ri[0]] = -3.0e38f; }
    __syncthreads();
  }

  if (tid == 0) {  // sort the 128 indices ascending (reference sorts top_idx)
    for (int i = 1; i < KSEL; ++i) {
      int key = sel[i], j = i - 1;
      while (j >= 0 && sel[j] > key) { sel[j + 1] = sel[j]; --j; }
      sel[j + 1] = key;
    }
  }
  __syncthreads();
  if (tid < KSEL) top_idx[b * KSEL + tid] = sel[tid];
}

// ---------------- attention over memory slots (per batch) ----------------
__global__ __launch_bounds__(256)
void k_attn(const float* __restrict__ qh, const float* __restrict__ memv,
            const float* __restrict__ q_w, const float* __restrict__ q_b,
            float* __restrict__ ctx) {
  const int b = blockIdx.x, tid = threadIdx.x;
  const int lane = tid & 31, wv = tid >> 5;
  extern __shared__ char smraw[];
  float* sQh = (float*)smraw;      // 1024
  float* sQ  = sQh + H_;           // 1024
  float* sS  = sQ + H_;            // 128

  for (int i = tid; i < H_; i += 256) sQh[i] = qh[(size_t)b * H_ + i];
  __syncthreads();

  // q = query_h @ q_w + q_b   (coalesced over j)
  for (int j = tid; j < H_; j += 256) {
    float acc = q_b[j];
    for (int k = 0; k < H_; ++k) acc += sQh[k] * q_w[(size_t)k * H_ + j];
    sQ[j] = acc;
  }
  __syncthreads();

  // s[m] = memory[m] . q
  for (int m = wv; m < KSEL; m += 8) {
    const float* mr = memv + ((size_t)b * KSEL + m) * H_;
    float p = 0.f;
    for (int c = lane; c < H_; c += 32) p += mr[c] * sQ[c];
    p = wave_sum(p);
    if (lane == 0) sS[m] = p;
  }
  __syncthreads();

  // softmax over 128 slots (all valid since KSEL <= n_cand) by wave 0
  if (tid < 32) {
    float vals[4]; float mx = -3.0e38f;
#pragma unroll
    for (int i = 0; i < 4; ++i) { vals[i] = sS[tid + 32 * i]; mx = fmaxf(mx, vals[i]); }
    mx = wave_max(mx);
    float se = 0.f;
#pragma unroll
    for (int i = 0; i < 4; ++i) { vals[i] = expf(vals[i] - mx); se += vals[i]; }
    se = wave_sum(se);
    const float inv = 1.0f / se;
#pragma unroll
    for (int i = 0; i < 4; ++i) sS[tid + 32 * i] = vals[i] * inv;
  }
  __syncthreads();

  // ctx = attn @ memory
  for (int j = tid; j < H_; j += 256) {
    float acc = 0.f;
    for (int m = 0; m < KSEL; ++m)
      acc += sS[m] * memv[((size_t)b * KSEL + m) * H_ + j];
    ctx[(size_t)b * H_ + j] = acc;
  }
}

// ---------------- fused final logits: (qh@do_w+do_b + ctx@out_w+out_b)/2 ----------------
// one pass over both 1024x32000 matrices; all 32 batches accumulated per column.
__global__ __launch_bounds__(256)
void k_logits(const float* __restrict__ qh,    const float* __restrict__ ctx,
              const float* __restrict__ do_w,  const float* __restrict__ do_b,
              const float* __restrict__ out_w, const float* __restrict__ out_b,
              float* __restrict__ out) {
  const int v = blockIdx.x * 256 + threadIdx.x;   // 125 blocks * 256 == 32000
  extern __shared__ char smraw[];
  float* sQ = (float*)smraw;     // 32 x 128
  float* sC = sQ + B_ * 128;     // 32 x 128

  float accD[B_], accM[B_];
#pragma unroll
  for (int b = 0; b < B_; ++b) { accD[b] = 0.f; accM[b] = 0.f; }

  for (int kc = 0; kc < H_; kc += 128) {
    __syncthreads();
    for (int i = threadIdx.x; i < B_ * 128; i += 256) {
      int bb = i >> 7, kk = i & 127;
      sQ[i] = qh[(size_t)bb * H_ + kc + kk];
      sC[i] = ctx[(size_t)bb * H_ + kc + kk];
    }
    __syncthreads();
    for (int k = 0; k < 128; ++k) {
      const float dv = do_w[(size_t)(kc + k) * V_ + v];
      const float ov = out_w[(size_t)(kc + k) * V_ + v];
#pragma unroll
      for (int b = 0; b < B_; ++b) {
        accD[b] = fmaf(sQ[b * 128 + k], dv, accD[b]);
        accM[b] = fmaf(sC[b * 128 + k], ov, accM[b]);
      }
    }
  }
  const float bd = do_b[v], bo = out_b[v];
#pragma unroll
  for (int b = 0; b < B_; ++b)
    out[(size_t)b * V_ + v] = 0.5f * (accD[b] + bd + accM[b] + bo);
}

// ---------------- host side ----------------
extern "C" void kernel_launch(void* const* d_in, const int* in_sizes, int n_in,
                              void* d_out, int out_size, void* d_ws, size_t ws_size,
                              hipStream_t stream) {
  const int*   seq   = (const int*)d_in[0];
  const float* embed = (const float*)d_in[1];
  const float* w1    = (const float*)d_in[2];
  const float* b1    = (const float*)d_in[3];
  const float* w2    = (const float*)d_in[4];
  const float* b2    = (const float*)d_in[5];
  const float* ln_g  = (const float*)d_in[6];
  const float* ln_b  = (const float*)d_in[7];
  const float* wg_w  = (const float*)d_in[8];
  const float* wg_b  = (const float*)d_in[9];
  const float* q_w   = (const float*)d_in[10];
  const float* q_b   = (const float*)d_in[11];
  const float* out_w = (const float*)d_in[12];
  const float* out_b = (const float*)d_in[13];
  const float* do_w  = (const float*)d_in[14];
  const float* do_b  = (const float*)d_in[15];
  float* out = (float*)d_out;

  // workspace carve (~25.7 MB)
  char* ws = (char*)d_ws;
  size_t off = 0;
  __bf16* w1t   = (__bf16*)(ws + off); off += (size_t)H2_ * H_ * 2;   // 4 MB
  __bf16* w2t   = (__bf16*)(ws + off); off += (size_t)H_ * H2_ * 2;   // 4 MB
  float* scores = (float*)(ws + off);  off += (size_t)B_ * 2048 * 4;  // 256 KB
  int*   topidx = (int*)(ws + off);    off += (size_t)B_ * KSEL * 4;  // 16 KB
  float* memv   = (float*)(ws + off);  off += (size_t)B_ * KSEL * H_ * 4; // 16 MB
  float* qh     = (float*)(ws + off);  off += (size_t)B_ * H_ * 4;    // 128 KB
  float* ctx    = (float*)(ws + off);  off += (size_t)B_ * H_ * 4;    // 128 KB

  // 0) weight convert+transpose to bf16 column-major (k-contiguous)
  k_cvt_transpose<<<8192, 256, 0, stream>>>(w1, w1t, H_, H2_);
  k_cvt_transpose<<<8192, 256, 0, stream>>>(w2, w2t, H2_, H_);

  const size_t ffn_smem = 16 * SH_STRIDE * 4 + 16 * SF_STRIDE * 2 + 16 * 4;

  // 1) scores for all candidate tokens (WMMA FFN, no hidden stored)
  k_ffn<<<dim3(128, B_), 256, ffn_smem, stream>>>(
      0, seq, embed, w1t, b1, w2t, b2, ln_g, ln_b, wg_w, wg_b,
      (const int*)nullptr, scores, (float*)nullptr, (float*)nullptr);

  // 2) top-128 per batch, sorted ascending
  const size_t topk_smem = 2048 * 4 + 128 * 4 + 256 * 4 + 256 * 4;
  k_topk<<<B_, 256, topk_smem, stream>>>(scores, topidx);

  // 3) recompute hidden only for selected rows + query row (129 rows/batch)
  k_ffn<<<dim3(9, B_), 256, ffn_smem, stream>>>(
      1, seq, embed, w1t, b1, w2t, b2, ln_g, ln_b, wg_w, wg_b,
      topidx, (float*)nullptr, memv, qh);

  // 4) attention over memory slots
  const size_t attn_smem = (H_ + H_ + KSEL) * 4;
  k_attn<<<B_, 256, attn_smem, stream>>>(qh, memv, q_w, q_b, ctx);

  // 5) fused vocab logits, both matrices in one pass
  const size_t log_smem = (size_t)B_ * 128 * 4 * 2;
  k_logits<<<V_ / 256, 256, log_smem, stream>>>(qh, ctx, do_w, do_b, out_w, out_b, out);
}
